// Encoding_43404939493633
// MI455X (gfx1250) — compile-verified, matrix operand order checked by reading, not run on hardware
//
#include <hip/hip_runtime.h>
#include <hip/hip_bf16.h>
#include <math.h>

// ---------------------------------------------------------------------------
// GNN encoder for MI455X (gfx1250, wave32).
//   h = l2norm(relu(X @ W1))
//   3x: hnv = scatter_add over edges; hnvs = scatter_add over batch;
//       h = l2norm(relu(h @ (W2@linW_top) + hnv @ (W3@linW_bot) + b))
// Dense update uses V_WMMA_F32_16X16X32_BF16 with fp32 accumulation.
// ---------------------------------------------------------------------------

typedef __attribute__((ext_vector_type(16))) __bf16 v16bf;
typedef __attribute__((ext_vector_type(8)))  float  v8f;

union BfPack {
    v16bf v;
    unsigned short s[16];
    uint4 q[2];
};

__device__ __forceinline__ unsigned short f2bf_rne(float f) {
    unsigned u = __float_as_uint(f);
    u += 0x7FFFu + ((u >> 16) & 1u);   // round-to-nearest-even (setup path only)
    return (unsigned short)(u >> 16);
}

// ---------------------------------------------------------------------------
// K1: pack combined weights A = W2@linW[0:64,:], Bm = W3@linW[64:128,:]
// directly into the per-lane WMMA B-operand layout (bf16, 32x16 K x N tiles).
// packed element index: (((mat*4 + tN)*2 + kHalf)*32 + lane)*16 + idx
// ---------------------------------------------------------------------------
__global__ void pack_weights(const float* __restrict__ W2,
                             const float* __restrict__ W3,
                             const float* __restrict__ linW,
                             unsigned short* __restrict__ packed) {
    int gid = blockIdx.x * blockDim.x + threadIdx.x;   // 0..8191
    if (gid >= 2 * 4 * 2 * 32 * 16) return;
    int idx  =  gid        & 15;
    int lane = (gid >> 4)  & 31;
    int kh   = (gid >> 9)  & 1;
    int tN   = (gid >> 10) & 3;
    int mat  = (gid >> 12) & 1;

    int laneM  = lane & 15;
    int laneHi = (lane >> 4) * 8;
    int Ncol = laneM + 16 * tN;
    int K = 32 * kh + ((idx < 8) ? (laneHi + idx) : (16 + laneHi + (idx - 8)));

    const float* Wm = mat ? W3 : W2;                 // [64][64] row-major
    const float* Lw = linW + (size_t)mat * 64 * 64;  // top / bottom half of [128][64]
    float acc = 0.f;
    #pragma unroll 8
    for (int j = 0; j < 64; ++j)
        acc += Wm[K * 64 + j] * Lw[j * 64 + Ncol];
    packed[gid] = f2bf_rne(acc);
}

// ---------------------------------------------------------------------------
// K2: h = l2norm(relu(X @ W1)),  X:[n,2], W1:[2,64]
// ---------------------------------------------------------------------------
__global__ void init_embed(const float* __restrict__ X,
                           const float* __restrict__ W1,
                           float* __restrict__ H, int n) {
    int i = blockIdx.x * blockDim.x + threadIdx.x;
    if (i >= n) return;
    float x0 = X[i * 2], x1 = X[i * 2 + 1];
    float v[64];
    float ss = 0.f;
    #pragma unroll
    for (int d = 0; d < 64; ++d) {
        float t = fmaxf(fmaf(x0, W1[d], x1 * W1[64 + d]), 0.f);
        v[d] = t;
        ss = fmaf(t, t, ss);
    }
    float rs = 1.f / fmaxf(sqrtf(ss), 1e-12f);
    #pragma unroll
    for (int d = 0; d < 64; ++d)
        H[(size_t)i * 64 + d] = v[d] * rs;
}

// ---------------------------------------------------------------------------
// K3: edge scatter-add: hnv[row] += h[col].  One wave per edge; lane owns 2
// floats -> coalesced 256B gather + global_atomic_add_f32 (no-return, agent).
// ---------------------------------------------------------------------------
__global__ void edge_scatter(const int* __restrict__ erow,
                             const int* __restrict__ ecol,
                             const float* __restrict__ h,
                             float* __restrict__ hnv, int nE) {
    int wave = (blockIdx.x * blockDim.x + threadIdx.x) >> 5;
    int lane = threadIdx.x & 31;
    if (wave >= nE) return;
    int r = erow[wave];
    int c = ecol[wave];
    float2 v = *(const float2*)(h + (size_t)c * 64 + lane * 2);
    float* dst = hnv + (size_t)r * 64 + lane * 2;
    __hip_atomic_fetch_add(dst,     v.x, __ATOMIC_RELAXED, __HIP_MEMORY_SCOPE_AGENT);
    __hip_atomic_fetch_add(dst + 1, v.y, __ATOMIC_RELAXED, __HIP_MEMORY_SCOPE_AGENT);
}

// ---------------------------------------------------------------------------
// K4: batch scatter-add with LDS staging: snv[bid[v]] += h[v].
// 16KB LDS accumulator (ds_add_f32), one global atomic flush per block.
// ---------------------------------------------------------------------------
__global__ void batch_scatter(const int* __restrict__ bids,
                              const float* __restrict__ h,
                              float* __restrict__ snv, int n) {
    __shared__ float acc[64 * 64];
    for (int i = threadIdx.x; i < 64 * 64; i += blockDim.x) acc[i] = 0.f;
    __syncthreads();
    int lane  = threadIdx.x & 31;
    int wave  = (blockIdx.x * blockDim.x + threadIdx.x) >> 5;
    int nWave = (gridDim.x * blockDim.x) >> 5;
    for (int v = wave; v < n; v += nWave) {
        int b = bids[v];
        float2 x = *(const float2*)(h + (size_t)v * 64 + lane * 2);
        float* d = &acc[b * 64 + lane * 2];
        __hip_atomic_fetch_add(d,     x.x, __ATOMIC_RELAXED, __HIP_MEMORY_SCOPE_WORKGROUP);
        __hip_atomic_fetch_add(d + 1, x.y, __ATOMIC_RELAXED, __HIP_MEMORY_SCOPE_WORKGROUP);
    }
    __syncthreads();
    for (int i = threadIdx.x; i < 64 * 64; i += blockDim.x)
        __hip_atomic_fetch_add(&snv[i], acc[i], __ATOMIC_RELAXED, __HIP_MEMORY_SCOPE_AGENT);
}

// ---------------------------------------------------------------------------
// K5: fused update: out = l2norm(relu(h@A + hnv@Bm + bias))
// One wave per 16-row tile; 16x v_wmma_f32_16x16x32_bf16 per tile.
// B-operand loads for tile t+1 are prefetched during tile t's WMMAs.
// ---------------------------------------------------------------------------
__device__ __forceinline__ v16bf load_a_bf16(const float* rowp, int kOff, int laneHi8) {
    // 16-bit A 16x32 layout: lanes 0-15 hold K = [kOff..kOff+8) and [kOff+16..kOff+24);
    // lanes 16-31 the +8 variants. Native f32->bf16 conversion (v_cvt).
    const float4* f0 = (const float4*)(rowp + kOff + laneHi8);
    const float4* f1 = (const float4*)(rowp + kOff + 16 + laneHi8);
    float4 x0 = f0[0], x1 = f0[1];
    float4 y0 = f1[0], y1 = f1[1];
    return (v16bf){
        (__bf16)x0.x, (__bf16)x0.y, (__bf16)x0.z, (__bf16)x0.w,
        (__bf16)x1.x, (__bf16)x1.y, (__bf16)x1.z, (__bf16)x1.w,
        (__bf16)y0.x, (__bf16)y0.y, (__bf16)y0.z, (__bf16)y0.w,
        (__bf16)y1.x, (__bf16)y1.y, (__bf16)y1.z, (__bf16)y1.w};
}

__device__ __forceinline__ v16bf load_b_packed(const uint4* packed, int mat, int tN,
                                               int kh, int lane) {
    BfPack p;
    int slot = ((mat * 4 + tN) * 2 + kh) * 32 + lane;   // 32 bytes per slot
    p.q[0] = packed[slot * 2];
    p.q[1] = packed[slot * 2 + 1];
    return p.v;
}

__global__ void fused_update(const float* __restrict__ h_in,
                             const float* __restrict__ h_nv,
                             const uint4* __restrict__ packedW,
                             const float* __restrict__ bias,
                             float* __restrict__ h_out,
                             int numTiles, int nRows) {
    int wave = (blockIdx.x * blockDim.x + threadIdx.x) >> 5;
    int lane = threadIdx.x & 31;
    if (wave >= numTiles) return;              // wave-uniform: EXEC stays all-ones
    int rowBase = wave * 16;
    int laneM   = lane & 15;
    int laneHi8 = (lane >> 4) * 8;

    int loadRow = rowBase + laneM;
    if (loadRow >= nRows) loadRow = nRows - 1; // clamp (keeps EXEC full)
    const float* rowA = h_in + (size_t)loadRow * 64;
    const float* rowG = h_nv + (size_t)loadRow * 64;

    v16bf a0 = load_a_bf16(rowA, 0,  laneHi8);
    v16bf a1 = load_a_bf16(rowA, 32, laneHi8);
    v16bf g0 = load_a_bf16(rowG, 0,  laneHi8);
    v16bf g1 = load_a_bf16(rowG, 32, laneHi8);

    // Software pipeline over the 4 column tiles.
    v16bf bA0 = load_b_packed(packedW, 0, 0, 0, lane);
    v16bf bA1 = load_b_packed(packedW, 0, 0, 1, lane);
    v16bf bB0 = load_b_packed(packedW, 1, 0, 0, lane);
    v16bf bB1 = load_b_packed(packedW, 1, 0, 1, lane);

    v8f c[4];
    #pragma unroll
    for (int t = 0; t < 4; ++t) {
        v16bf nA0, nA1, nB0, nB1;
        if (t < 3) {
            nA0 = load_b_packed(packedW, 0, t + 1, 0, lane);
            nA1 = load_b_packed(packedW, 0, t + 1, 1, lane);
            nB0 = load_b_packed(packedW, 1, t + 1, 0, lane);
            nB1 = load_b_packed(packedW, 1, t + 1, 1, lane);
        }
        v8f acc = {};
        acc = __builtin_amdgcn_wmma_f32_16x16x32_bf16(false, a0, false, bA0,
                                                      (short)0, acc, false, false);
        acc = __builtin_amdgcn_wmma_f32_16x16x32_bf16(false, a1, false, bA1,
                                                      (short)0, acc, false, false);
        acc = __builtin_amdgcn_wmma_f32_16x16x32_bf16(false, g0, false, bB0,
                                                      (short)0, acc, false, false);
        acc = __builtin_amdgcn_wmma_f32_16x16x32_bf16(false, g1, false, bB1,
                                                      (short)0, acc, false, false);
        float bcol = bias[laneM + 16 * t];
        #pragma unroll
        for (int r = 0; r < 8; ++r)
            acc[r] = fmaxf(acc[r] + bcol, 0.f);
        c[t] = acc;
        if (t < 3) { bA0 = nA0; bA1 = nA1; bB0 = nB0; bB1 = nB1; }
    }

    // Row-wise L2 norm: row M = laneHi8 + r lives across 16 lanes of this half.
    float rs[8];
    #pragma unroll
    for (int r = 0; r < 8; ++r) {
        float s = 0.f;
        #pragma unroll
        for (int t = 0; t < 4; ++t) s = fmaf(c[t][r], c[t][r], s);
        #pragma unroll
        for (int m = 1; m <= 8; m <<= 1) s += __shfl_xor(s, m);  // stays in 16-lane half
        rs[r] = 1.f / fmaxf(sqrtf(s), 1e-12f);
    }

    #pragma unroll
    for (int r = 0; r < 8; ++r) {
        int row = rowBase + laneHi8 + r;
        if (row < nRows) {
            #pragma unroll
            for (int t = 0; t < 4; ++t)
                h_out[(size_t)row * 64 + laneM + 16 * t] = c[t][r] * rs[r];
        }
    }
}

// ---------------------------------------------------------------------------
extern "C" void kernel_launch(void* const* d_in, const int* in_sizes, int n_in,
                              void* d_out, int out_size, void* d_ws, size_t ws_size,
                              hipStream_t stream) {
    const int*   erow = (const int*)d_in[0];
    const int*   ecol = (const int*)d_in[1];
    const int*   bids = (const int*)d_in[2];
    const float* X    = (const float*)d_in[3];
    const float* Xs   = (const float*)d_in[4];
    const float* W1   = (const float*)d_in[5];
    const float* W2   = (const float*)d_in[6];
    const float* W3   = (const float*)d_in[7];
    const float* linW = (const float*)d_in[8];
    const float* linb = (const float*)d_in[9];
    const int nE = in_sizes[0];
    const int nN = in_sizes[2];
    const int nB = in_sizes[4] / 2;        // D_IN = 2
    const int depth = 3;                   // reference DEPTH (compile-time constant)

    float* ws = (float*)d_ws;
    size_t nd = (size_t)nN * 64;
    size_t bd = (size_t)nB * 64;
    float* bufA = ws;
    float* bufB = bufA + nd;
    float* hnv  = bufB + nd;
    float* sA   = hnv + nd;
    float* sB   = sA + bd;
    float* snv  = sB + bd;
    unsigned short* packed = (unsigned short*)(snv + bd);   // 16 KB, 16B-aligned

    hipLaunchKernelGGL(pack_weights, dim3(32), dim3(256), 0, stream, W2, W3, linW, packed);
    hipLaunchKernelGGL(init_embed, dim3((nN + 255) / 256), dim3(256), 0, stream, X,  W1, bufA, nN);
    hipLaunchKernelGGL(init_embed, dim3((nB + 255) / 256), dim3(256), 0, stream, Xs, W1, sA,  nB);

    float* outN = (float*)d_out;
    float* outS = outN + nd;
    float* hin = bufA; float* hfree = bufB;
    float* sin = sA;   float* sfree = sB;
    int tilesN = (nN + 15) / 16;
    int tilesB = (nB + 15) / 16;

    for (int it = 0; it < depth; ++it) {
        hipMemsetAsync(hnv, 0, nd * sizeof(float), stream);
        hipMemsetAsync(snv, 0, bd * sizeof(float), stream);

        int eb = (int)(((size_t)nE * 32 + 255) / 256);
        hipLaunchKernelGGL(edge_scatter, dim3(eb), dim3(256), 0, stream,
                           erow, ecol, hin, hnv, nE);
        hipLaunchKernelGGL(batch_scatter, dim3(128), dim3(256), 0, stream,
                           bids, hin, snv, nN);

        float* hout = (it == depth - 1) ? outN : hfree;
        float* sout = (it == depth - 1) ? outS : sfree;
        hipLaunchKernelGGL(fused_update, dim3((tilesN * 32 + 255) / 256), dim3(256), 0,
                           stream, hin, hnv, (const uint4*)packed, linb, hout, tilesN, nN);
        hipLaunchKernelGGL(fused_update, dim3((tilesB * 32 + 255) / 256), dim3(256), 0,
                           stream, sin, snv, (const uint4*)packed, linb, sout, tilesB, nB);

        float* t = hin; hin = hout; hfree = t;
        t = sin; sin = sout; sfree = t;
    }
    (void)n_in; (void)out_size; (void)ws_size;
}
